// AttentionPooling_27616639713402
// MI455X (gfx1250) — matrix operand order repaired and checked by reading, not compiled, not run on hardware
//
#include <hip/hip_runtime.h>

// ---------------------------------------------------------------------------
// AttentionPooling for MI455X (gfx1250, wave32, WMMA bf16 16x16x32)
//
// Math reduction: out[b,o] = sum_p softmax_col(logits)[o,p] * vmean[b,p]
//   logits[b]  = A[b] + bq*Sk^T + Sq*bk^T + N*bq*bk^T
//   A[b]       = (Wq X_b)(Wk X_b)^T          (contract c=512, n=4096)
//   S*[b]      = W* @ xsum[b],  vmean = Sv/N + bv
// Only A and xsum touch the 256MB x tensor -> single streaming pass + WMMA.
//
// All WMMA operand fragments are contiguous 2x16B ds_load_b128 per lane
// (x tile stored pixel-major); LDS rows padded so row-strided b128 fragment
// reads tile all 64 banks. Global x loads are software-pipelined through a
// float4[8] register buffer: each phase issues the NEXT phase's 8x b128
// clause before the barrier, hiding HBM latency under the WMMA phase
// (we run 1 WG/WGP due to 156KB LDS, so there is no inter-WG overlap).
// ---------------------------------------------------------------------------

typedef __bf16 bf16_t;
typedef __attribute__((ext_vector_type(16))) __bf16 v16bf;
typedef __attribute__((ext_vector_type(8)))  __bf16 v8bf;
typedef __attribute__((ext_vector_type(8)))  float  v8f;

#define BATCHES 32
#define CIN     512
#define C8      64
#define NPIX    4096
#define NCHUNK  8                      // workgroups per batch along N
#define PIX_PER_WG (NPIX / NCHUNK)     // 512
#define NT      32                     // pixels per subtile (= WMMA K of gemm2)
#define NSUB    (PIX_PER_WG / NT)      // 16
#define NPHASE  (2 * NSUB)             // 32 half-tile load/compute phases

#define WPAD    520                    // weight row pitch (512+8) -> 4-bank stride
#define XPAD    264                    // x-tile row pitch (256+8)
#define QKPAD   40                     // q/k row pitch (32+8)

union U16bf { v16bf v; v8bf h[2]; };

// A-matrix fragment (16x32 bf16, MxK). Lane l: m = l&15, half = l>>4.
// v0..v3 hold K = k0+[8h..8h+7], v4..v7 hold K = k0+[16+8h..23+8h]
// -> two contiguous 16B LDS reads per lane.
__device__ __forceinline__ v16bf lds_afrag(const bf16_t* base, int row0,
                                           int stride, int k0) {
  const int lane = threadIdx.x & 31;
  const int h = lane >> 4, m = lane & 15;
  const bf16_t* p = base + (row0 + m) * stride + k0 + 8 * h;
  U16bf u;
  u.h[0] = *(const v8bf*)(p);
  u.h[1] = *(const v8bf*)(p + 16);
  return u.v;
}

// B-matrix fragment (32x16 bf16, KxN) where B[k][n] = base[(col0+n)*stride+k0+k]
// (stored matrix is N-major with contiguous K). Lane l: n = l&15, h = l>>4,
// e0..7 hold K = k0+[8h..8h+7], e8..15 hold K = k0+[16+8h..23+8h] -> two 16B LDS reads per lane.
__device__ __forceinline__ v16bf lds_bfrag(const bf16_t* base, int col0,
                                           int stride, int k0) {
  const int lane = threadIdx.x & 31;
  const int h = lane >> 4, n = lane & 15;
  const bf16_t* p = base + (col0 + n) * stride + k0 + 8 * h;
  U16bf u;
  u.h[0] = *(const v8bf*)(p);
  u.h[1] = *(const v8bf*)(p + 16);
  return u.v;
}

__global__ void zero_ws_kernel(float* __restrict__ p, int n) {
  int i = blockIdx.x * 256 + threadIdx.x;
  if (i < n) p[i] = 0.0f;
}

// ---------------------------------------------------------------------------
// Main pass: per (batch, n-chunk) workgroup, accumulate A[b] and xsum[b]
// (global f32 atomics) from a single streaming read of x.
// LDS: wq 65KB + wk 65KB + xT 16.5KB + q 5KB + k 5KB = 156.5KB (<320KB WGP).
// ---------------------------------------------------------------------------
__global__ __launch_bounds__(256) void attn_main_kernel(
    const float* __restrict__ x,   // [32,512,4096]
    const float* __restrict__ wq,  // [64,512]
    const float* __restrict__ wk,  // [64,512]
    float* __restrict__ Abuf,      // [32,64,64] accumulator
    float* __restrict__ xsum) {    // [32,512]  accumulator
  __shared__ alignas(16) bf16_t wq_s[C8 * WPAD];
  __shared__ alignas(16) bf16_t wk_s[C8 * WPAD];
  __shared__ alignas(16) bf16_t x_s[NT * XPAD];  // pixel-major half-tile
  __shared__ alignas(16) bf16_t q_s[C8 * QKPAD];
  __shared__ alignas(16) bf16_t k_s[C8 * QKPAD];

  const int tid   = threadIdx.x;
  const int lane  = tid & 31;
  const int wave  = tid >> 5;          // 0..7
  const int b     = blockIdx.y;
  const int chunk = blockIdx.x;

  // Prefetch helper: issue the 8x global_load_b128 clause for phase `ph`
  // (phase = subtile*2 + channel-half). The s_wait_loadcnt is sunk by the
  // compiler to where fbuf is consumed (next phase's convert), so the loads
  // fly across the barrier + the current phase's 16 WMMAs.
  auto issue_loads = [&](int ph, float4* fb) {
    const int c  = ((ph & 1) << 8) + tid;              // this thread's channel
    const int n0 = chunk * PIX_PER_WG + (ph >> 1) * NT;
    const float4* xr =
        reinterpret_cast<const float4*>(x + ((size_t)(b * CIN + c)) * NPIX + n0);
#pragma unroll
    for (int j = 0; j < 8; ++j) fb[j] = xr[j];
  };

  float4 fbuf[8];
  issue_loads(0, fbuf);                // x loads in flight during weight staging

  // Stage weights fp32 -> bf16 into LDS (reused for all 512 pixels).
  for (int i = tid; i < (C8 * CIN) / 4; i += 256) {
    float4 fq = reinterpret_cast<const float4*>(wq)[i];
    float4 fk = reinterpret_cast<const float4*>(wk)[i];
    const int row = (i * 4) >> 9;            // /512
    const int col = (i * 4) & 511;
    bf16_t* dq = &wq_s[row * WPAD + col];
    bf16_t* dk = &wk_s[row * WPAD + col];
    dq[0] = (bf16_t)fq.x; dq[1] = (bf16_t)fq.y;
    dq[2] = (bf16_t)fq.z; dq[3] = (bf16_t)fq.w;
    dk[0] = (bf16_t)fk.x; dk[1] = (bf16_t)fk.y;
    dk[2] = (bf16_t)fk.z; dk[3] = (bf16_t)fk.w;
  }
  __syncthreads();

  const int oB  = (wave >> 1) * 16;    // output-channel tile (q,k,gemm2 rows)
  const int nB  = (wave & 1) * 16;     // pixel tile in projection
  const int pB0 = (wave & 1) * 32;     // gemm2 column base (2 tiles per wave)

  v8f acc2[2];
  acc2[0] = v8f{0.f, 0.f, 0.f, 0.f, 0.f, 0.f, 0.f, 0.f};
  acc2[1] = v8f{0.f, 0.f, 0.f, 0.f, 0.f, 0.f, 0.f, 0.f};
  v8f qacc, kacc;

  float rs0 = 0.f, rs1 = 0.f;          // per-thread channel sums (c=tid, tid+256)

  for (int ph = 0; ph < NPHASE; ++ph) {
    const int half = ph & 1;
    if (half == 0) {
      qacc = v8f{0.f, 0.f, 0.f, 0.f, 0.f, 0.f, 0.f, 0.f};
      kacc = v8f{0.f, 0.f, 0.f, 0.f, 0.f, 0.f, 0.f, 0.f};
    }

    // Convert prefetched 256ch x 32px block, store TRANSPOSED (pixel-major)
    // as bf16, folding per-channel running sums for xsum.
    {
      float s = 0.f;
#pragma unroll
      for (int j = 0; j < 8; ++j) {
        const float4 f = fbuf[j];
        s += f.x + f.y + f.z + f.w;
        x_s[(j * 4 + 0) * XPAD + tid] = (bf16_t)f.x;
        x_s[(j * 4 + 1) * XPAD + tid] = (bf16_t)f.y;
        x_s[(j * 4 + 2) * XPAD + tid] = (bf16_t)f.z;
        x_s[(j * 4 + 3) * XPAD + tid] = (bf16_t)f.w;
      }
      if (half == 0) rs0 += s; else rs1 += s;
    }
    if (ph + 1 < NPHASE) issue_loads(ph + 1, fbuf);   // prefetch next phase
    __syncthreads();                                   // (A) x_s visible

    // Projection GEMMs: q_tile += Wq[oB:, kg:] @ x_t ; same for k.
    // B[k][n] = x_s[(nB+n)*XPAD + klocal] -> contiguous-K fragment.
#pragma unroll 4
    for (int kk = 0; kk < 8; ++kk) {
      const int kg = half * 256 + kk * 32;
      v16bf aq = lds_afrag(wq_s, oB, WPAD, kg);
      v16bf ak = lds_afrag(wk_s, oB, WPAD, kg);
      v16bf bx = lds_bfrag(x_s, nB, XPAD, kk * 32);
      qacc = __builtin_amdgcn_wmma_f32_16x16x32_bf16(
          false, aq, false, bx, (short)0, qacc, false, false);
      kacc = __builtin_amdgcn_wmma_f32_16x16x32_bf16(
          false, ak, false, bx, (short)0, kacc, false, false);
    }

    if (half == 1) {
      // Round q/k tiles to bf16 in LDS (C/D layout: VGPR j -> row j+8*half).
      const int h = lane >> 4, ln = lane & 15;
#pragma unroll
      for (int j = 0; j < 8; ++j) {
        q_s[(oB + j + 8 * h) * QKPAD + nB + ln] = (bf16_t)qacc[j];
        k_s[(oB + j + 8 * h) * QKPAD + nB + ln] = (bf16_t)kacc[j];
      }
      __syncthreads();                 // (C) q_s/k_s visible to all waves

      // A += q @ k^T over this subtile's 32 pixels (K = 32 -> one WMMA/tile).
      v16bf aq2 = lds_afrag(q_s, oB, QKPAD, 0);
      v16bf bk0 = lds_bfrag(k_s, pB0, QKPAD, 0);
      v16bf bk1 = lds_bfrag(k_s, pB0 + 16, QKPAD, 0);
      acc2[0] = __builtin_amdgcn_wmma_f32_16x16x32_bf16(
          false, aq2, false, bk0, (short)0, acc2[0], false, false);
      acc2[1] = __builtin_amdgcn_wmma_f32_16x16x32_bf16(
          false, aq2, false, bk1, (short)0, acc2[1], false, false);
    }
    __syncthreads();                   // (D) protect x_s (and q_s/k_s reads)
  }

  // Commit partial A[b] (8 chunks per batch contend) and xsum[b].
  {
    const int h = lane >> 4, ln = lane & 15;
    float* Ab = Abuf + b * (C8 * C8);
#pragma unroll
    for (int t = 0; t < 2; ++t) {
      const int pB = pB0 + t * 16;
#pragma unroll
      for (int j = 0; j < 8; ++j) {
        atomicAdd(&Ab[(oB + j + 8 * h) * C8 + pB + ln], acc2[t][j]);
      }
    }
  }
  atomicAdd(&xsum[b * CIN + tid], rs0);
  atomicAdd(&xsum[b * CIN + 256 + tid], rs1);
}

// ---------------------------------------------------------------------------
// Epilogue: biases, column softmax (axis=-2), attn @ vmean. Tiny.
// ---------------------------------------------------------------------------
__global__ __launch_bounds__(64) void attn_finalize_kernel(
    const float* __restrict__ Abuf, const float* __restrict__ xsum,
    const float* __restrict__ wq, const float* __restrict__ bq,
    const float* __restrict__ wk, const float* __restrict__ bk,
    const float* __restrict__ wv, const float* __restrict__ bv,
    float* __restrict__ out) {
  __shared__ float xs[CIN];
  __shared__ float Sq[C8], Sk[C8], Vm[C8], Bq[C8], Bk[C8];
  __shared__ float wsm[C8][C8 + 1];   // padded vs bank conflicts
  const int b = blockIdx.x, t = threadIdx.x;

  for (int i = t; i < CIN; i += 64) xs[i] = xsum[b * CIN + i];
  __syncthreads();

  float sq = 0.f, sk = 0.f, sv = 0.f;
  for (int c = 0; c < CIN; ++c) {
    const float xc = xs[c];
    sq += wq[t * CIN + c] * xc;
    sk += wk[t * CIN + c] * xc;
    sv += wv[t * CIN + c] * xc;
  }
  Sq[t] = sq;
  Sk[t] = sk;
  Vm[t] = sv * (1.0f / (float)NPIX) + bv[t];
  Bq[t] = bq[t];
  Bk[t] = bk[t];
  __syncthreads();

  // Thread t owns column p=t of logits; softmax normalizes over o (axis=-2).
  const float* Ab = Abuf + b * C8 * C8;
  const float skt = Sk[t], bkt = Bk[t];
  float mx = -3.0e38f;
  for (int o = 0; o < C8; ++o) {
    const float l = Ab[o * C8 + t] + Bq[o] * skt + bkt * Sq[o] +
                    (float)NPIX * Bq[o] * bkt;
    wsm[o][t] = l;
    mx = fmaxf(mx, l);
  }
  float ssum = 0.f;
  for (int o = 0; o < C8; ++o) {
    const float e = __expf(wsm[o][t] - mx);
    wsm[o][t] = e;
    ssum += e;
  }
  const float wgt = Vm[t] / ssum;   // fold vmean[p] into column weights
  for (int o = 0; o < C8; ++o) wsm[o][t] *= wgt;
  __syncthreads();

  float r = 0.f;
  for (int p = 0; p < C8; ++p) r += wsm[t][p];
  out[b * C8 + t] = r;
}

extern "C" void kernel_launch(void* const* d_in, const int* in_sizes, int n_in,
                              void* d_out, int out_size, void* d_ws, size_t ws_size,
                              hipStream_t stream) {
  (void)in_sizes; (void)n_in; (void)out_size; (void)ws_size;
  const float* x  = (const float*)d_in[0];
  const float* wq = (const float*)d_in[1];
  const float* bq = (const float*)d_in[2];
  const float* wk = (const float*)d_in[3];
  const float* bk = (const float*)d_in[4];
  const float* wv = (const float*)d_in[5];
  const float* bv = (const float*)d_in[6];
  float* out  = (float*)d_out;
  float* Abuf = (float*)d_ws;                       // 32*64*64 floats
  float* xsum = Abuf + BATCHES * C8 * C8;           // 32*512 floats

  const int nz = BATCHES * C8 * C8 + BATCHES * CIN;
  zero_ws_kernel<<<(nz + 255) / 256, 256, 0, stream>>>(Abuf, nz);
  attn_main_kernel<<<dim3(NCHUNK, BATCHES), 256, 0, stream>>>(x, wq, wk, Abuf, xsum);
  attn_finalize_kernel<<<BATCHES, 64, 0, stream>>>(Abuf, xsum, wq, bq, wk, bk,
                                                   wv, bv, out);
}
